// MultiHeadAttentionV2_86586540687897
// MI455X (gfx1250) — matrix operand order repaired and checked by reading, not compiled, not run on hardware
//
#include <hip/hip_runtime.h>
#include <hip/hip_bf16.h>

#define SEQ 2048
#define DM  1024
#define DK  64
#define NH  16

typedef __attribute__((ext_vector_type(16))) __bf16 v16bf;
typedef __attribute__((ext_vector_type(8)))  float  v8f;

union BF16Frag {
  v16bf v;
  uint4 q[2];
  unsigned short s[16];
};

// fp32 -> bf16 scalar, round-to-nearest-even
__device__ __forceinline__ unsigned short f2bf(float f) {
  union { float f; unsigned u; } c; c.f = f;
  unsigned r = c.u + 0x7FFFu + ((c.u >> 16) & 1u);
  return (unsigned short)(r >> 16);
}

// pack two fp32 -> two bf16 (RNE) in one dword
__device__ __forceinline__ unsigned pk2bf(float lo, float hi) {
  union { float f; unsigned u; } a, b; a.f = lo; b.f = hi;
  unsigned ra = a.u + 0x7FFFu + ((a.u >> 16) & 1u);
  unsigned rb = b.u + 0x7FFFu + ((b.u >> 16) & 1u);
  return __builtin_amdgcn_perm(rb, ra, 0x07060302u);
}

__device__ __forceinline__ v8f wmma_bf16(const BF16Frag& a, const BF16Frag& b, v8f c) {
  return __builtin_amdgcn_wmma_f32_16x16x32_bf16(false, a.v, false, b.v,
                                                 (short)0, c, false, false);
}

// async copy 16 bytes global -> LDS (ASYNCcnt-tracked)
__device__ __forceinline__ void async_g2l_b128(unsigned lds_off, const void* gptr) {
  unsigned long long ga = (unsigned long long)(uintptr_t)gptr;
  asm volatile("global_load_async_to_lds_b128 %0, %1, off"
               :: "v"(lds_off), "v"(ga) : "memory");
}
__device__ __forceinline__ void wait_async0() {
  asm volatile("s_wait_asynccnt 0x0" ::: "memory");
}
__device__ __forceinline__ void wait_async2() {
  asm volatile("s_wait_asynccnt 0x2" ::: "memory");
}

// ---------------------------------------------------------------------------
// Bulk fp32 -> bf16 conversion (bandwidth-bound, done once per buffer).
// ---------------------------------------------------------------------------
__global__ __launch_bounds__(256)
void cvt_f32_bf16(const float* __restrict__ in, unsigned short* __restrict__ out, int n4)
{
  int i = blockIdx.x * blockDim.x + threadIdx.x;
  if (i >= n4) return;
  float4 f = ((const float4*)in)[i];
  uint2 o;
  o.x = pk2bf(f.x, f.y);
  o.y = pk2bf(f.z, f.w);
  ((uint2*)out)[i] = o;
}

// ---------------------------------------------------------------------------
// Cooperative GEMM: out = Xbf(2048x1024) @ Wbf^T(1024x1024) + bias.
// Block = 8 waves = 128x64 output tile. K-step 64 with DOUBLE-BUFFERED
// async Global->LDS staging of the 64x64 B tile (8 KB/buffer): the copy of
// tile i+1 is in flight while tile i feeds 8 WMMAs per wave from ds_load_b128.
// mode 0: bf16 head-split [H][S][64]  (Q, K)
// mode 1: bf16 transposed [H][64][S]  (V)
// mode 2: fp32 row-major  [S][1024]   (final output)
// ---------------------------------------------------------------------------
__global__ __launch_bounds__(256)
void gemm_bf16_xwT(const unsigned short* __restrict__ X,
                   const unsigned short* __restrict__ W,
                   const float* __restrict__ bias, void* __restrict__ out, int mode)
{
  __shared__ __align__(16) unsigned short Bt[2][64 * 64];   // 2 x 8 KB

  const int lane = threadIdx.x & 31;
  const int hl   = lane >> 4;
  const int l16  = lane & 15;
  const int w    = threadIdx.x >> 5;

  const int nTilesN = DM / 64;                              // 16
  const int m0 = (blockIdx.x / nTilesN) * 128 + w * 16;
  const int n0 = (blockIdx.x % nTilesN) * 64;

  // copy assignment: thread i moves 2 x 16B of row (i>>2), 32B chunk (i&3)
  const int crow = threadIdx.x >> 2;
  const int cch  = threadIdx.x & 3;
  const unsigned ldsBase =
      ((unsigned)(uintptr_t)(&Bt[0][0])) + (unsigned)(crow * 128 + cch * 32);
  const unsigned short* gsrc = W + (size_t)(n0 + crow) * DM + cch * 16;

  const unsigned short* xrow = X + (size_t)(m0 + l16) * DM;

  v8f C[4];
  #pragma unroll
  for (int t = 0; t < 4; t++) C[t] = (v8f){};

  // prologue: stage tile 0 into buffer 0
  async_g2l_b128(ldsBase,      gsrc);
  async_g2l_b128(ldsBase + 16, gsrc + 8);

  const int NK = DM / 64;                                   // 16 K-steps
  for (int i = 0; i < NK; i++) {
    const int k   = i * 64;
    const int cur = i & 1;

    if (i + 1 < NK) {   // stage next tile into the other buffer, then wait
      const unsigned dst = ldsBase + (unsigned)(((i + 1) & 1) << 13);
      async_g2l_b128(dst,      gsrc + k + 64);
      async_g2l_b128(dst + 16, gsrc + k + 64 + 8);
    }

    // A fragments for both K-halves while copies are in flight
    BF16Frag A0, A1;
    {
      const uint4* xa = (const uint4*)(xrow + k + hl * 16);
      A0.q[0] = xa[0];  A0.q[1] = xa[1];    // K = k    + hl*16 + 0..15
      A1.q[0] = xa[4];  A1.q[1] = xa[5];    // K = k+32 + hl*16 + 0..15
    }

    if (i + 1 < NK) wait_async2(); else wait_async0();
    __syncthreads();

    const unsigned short* Bc = &Bt[cur][0];
    #pragma unroll
    for (int kk = 0; kk < 2; kk++) {
      #pragma unroll
      for (int t = 0; t < 4; t++) {
        // B fragment: lane col = n0+t*16+l16, K = kk*32 + hl*16 + 0..15
        const uint4* bb = (const uint4*)(Bc + (t * 16 + l16) * 64 + kk * 32 + hl * 16);
        BF16Frag B;
        B.q[0] = bb[0];
        B.q[1] = bb[1];
        C[t] = wmma_bf16(kk ? A1 : A0, B, C[t]);
      }
    }
    __syncthreads();   // all reads of Bt[cur] done before it is overwritten
  }

  #pragma unroll
  for (int t = 0; t < 4; t++) {
    const int col  = n0 + t * 16 + l16;
    const float bv = bias[col];
    const int hidx = col >> 6;
    const int didx = col & 63;
    #pragma unroll
    for (int r = 0; r < 8; r++) {
      float cval = C[t][r] + bv;
      int m = m0 + r + hl * 8;
      if (mode == 0) {
        ((unsigned short*)out)[(size_t)hidx * SEQ * DK + (size_t)m * DK + didx] = f2bf(cval);
      } else if (mode == 1) {
        ((unsigned short*)out)[(size_t)hidx * DK * SEQ + (size_t)didx * SEQ + m] = f2bf(cval);
      } else {
        ((float*)out)[(size_t)m * DM + col] = cval;
      }
    }
  }
}

// ---------------------------------------------------------------------------
// Flash attention: one wave = one (head, 16-query tile). 4 waves/block share
// the same query tile (different heads) -> uniform trip count, safe barriers.
// Writes ctx as bf16 [S][D_MODEL] for the bf16 out-projection GEMM.
// ---------------------------------------------------------------------------
__global__ __launch_bounds__(128)
void attn_flash(const unsigned short* __restrict__ Qb,
                const unsigned short* __restrict__ Kb,
                const unsigned short* __restrict__ Vt,
                unsigned short* __restrict__ ctx)
{
  __shared__ __align__(16) unsigned short Pl[4][16 * 32];

  const int lane = threadIdx.x & 31;
  const int hl   = lane >> 4;
  const int l16  = lane & 15;
  const int w    = threadIdx.x >> 5;
  const int qt   = blockIdx.x >> 2;
  const int hg   = blockIdx.x & 3;
  const int h    = hg * 4 + w;
  const int q0   = qt * 16;
  unsigned short* myP = Pl[w];

  const unsigned short* Qh = Qb + (size_t)h * SEQ * DK;
  const unsigned short* Kh = Kb + (size_t)h * SEQ * DK;
  const unsigned short* Vh = Vt + (size_t)h * DK * SEQ;

  BF16Frag AQ0, AQ1;
  {
    const uint4* qp = (const uint4*)(Qh + (size_t)(q0 + l16) * DK + hl * 16);
    AQ0.q[0] = qp[0]; AQ0.q[1] = qp[1];
    AQ1.q[0] = qp[4]; AQ1.q[1] = qp[5];
  }

  float mrow[8], lrow[8];
  v8f C[4];
  #pragma unroll
  for (int r = 0; r < 8; r++) { mrow[r] = -1e30f; lrow[r] = 0.0f; }
  #pragma unroll
  for (int t = 0; t < 4; t++) C[t] = (v8f){};

  for (int kb0 = 0; kb0 < q0 + 16; kb0 += 32) {
    v8f s0 = {}, s1 = {};
    {
      const uint4* kp = (const uint4*)(Kh + (size_t)(kb0 + l16) * DK + hl * 16);
      BF16Frag BK;
      BK.q[0] = kp[0]; BK.q[1] = kp[1];
      s0 = wmma_bf16(AQ0, BK, s0);
      BK.q[0] = kp[4]; BK.q[1] = kp[5];
      s0 = wmma_bf16(AQ1, BK, s0);
    }
    {
      const uint4* kp = (const uint4*)(Kh + (size_t)(kb0 + 16 + l16) * DK + hl * 16);
      BF16Frag BK;
      BK.q[0] = kp[0]; BK.q[1] = kp[1];
      s1 = wmma_bf16(AQ0, BK, s1);
      BK.q[0] = kp[4]; BK.q[1] = kp[5];
      s1 = wmma_bf16(AQ1, BK, s1);
    }

    float p0[8], p1[8], mx[8];
    #pragma unroll
    for (int r = 0; r < 8; r++) {
      int qrow = q0 + r + hl * 8;
      float a = s0[r] * 0.125f;
      float b = s1[r] * 0.125f;
      if (kb0 + l16 > qrow)      a = -1e30f;
      if (kb0 + 16 + l16 > qrow) b = -1e30f;
      p0[r] = a; p1[r] = b;
      mx[r] = fmaxf(a, b);
    }
    #pragma unroll
    for (int off = 8; off; off >>= 1)
      #pragma unroll
      for (int r = 0; r < 8; r++)
        mx[r] = fmaxf(mx[r], __shfl_xor(mx[r], off, 32));

    float rs[8];
    #pragma unroll
    for (int r = 0; r < 8; r++) {
      float mnew  = fmaxf(mrow[r], mx[r]);
      float alpha = __expf(mrow[r] - mnew);
      mrow[r] = mnew;
      p0[r] = __expf(p0[r] - mnew);
      p1[r] = __expf(p1[r] - mnew);
      rs[r] = p0[r] + p1[r];
      lrow[r] *= alpha;
      #pragma unroll
      for (int t = 0; t < 4; t++) C[t][r] *= alpha;
    }
    #pragma unroll
    for (int off = 8; off; off >>= 1)
      #pragma unroll
      for (int r = 0; r < 8; r++)
        rs[r] += __shfl_xor(rs[r], off, 32);
    #pragma unroll
    for (int r = 0; r < 8; r++) lrow[r] += rs[r];

    #pragma unroll
    for (int r = 0; r < 8; r++) {
      int row = r + hl * 8;
      myP[row * 32 + l16]      = f2bf(p0[r]);
      myP[row * 32 + 16 + l16] = f2bf(p1[r]);
    }
    __syncthreads();
    BF16Frag AP;
    {
      const uint4* pp = (const uint4*)(myP + l16 * 32 + hl * 16);
      AP.q[0] = pp[0]; AP.q[1] = pp[1];
    }
    __syncthreads();

    #pragma unroll
    for (int t = 0; t < 4; t++) {
      const uint4* vp = (const uint4*)(Vh + (size_t)(t * 16 + l16) * SEQ + kb0 + hl * 16);
      BF16Frag BV;
      BV.q[0] = vp[0]; BV.q[1] = vp[1];
      C[t] = wmma_bf16(AP, BV, C[t]);
    }
  }

  #pragma unroll
  for (int r = 0; r < 8; r++) {
    float inv = 1.0f / lrow[r];
    int row = q0 + r + hl * 8;
    #pragma unroll
    for (int t = 0; t < 4; t++) {
      ctx[(size_t)row * DM + h * DK + t * 16 + l16] = f2bf(C[t][r] * inv);
    }
  }
}

// ---------------------------------------------------------------------------
extern "C" void kernel_launch(void* const* d_in, const int* in_sizes, int n_in,
                              void* d_out, int out_size, void* d_ws, size_t ws_size,
                              hipStream_t stream) {
  const float* Xq = (const float*)d_in[0];
  const float* Xk = (const float*)d_in[1];
  const float* Xv = (const float*)d_in[2];
  const float* Wq = (const float*)d_in[3];
  const float* bq = (const float*)d_in[4];
  const float* Wk = (const float*)d_in[5];
  const float* bk = (const float*)d_in[6];
  const float* Wv = (const float*)d_in[7];
  const float* bv = (const float*)d_in[8];
  const float* Wo = (const float*)d_in[9];
  const float* bo = (const float*)d_in[10];

  char* ws = (char*)d_ws;
  const size_t MB = 1u << 20;
  unsigned short* Xqb = (unsigned short*)(ws + 0 * MB);
  unsigned short* Xkb = (unsigned short*)(ws + 4 * MB);
  unsigned short* Xvb = (unsigned short*)(ws + 8 * MB);
  unsigned short* Wqb = (unsigned short*)(ws + 12 * MB);
  unsigned short* Wkb = (unsigned short*)(ws + 14 * MB);
  unsigned short* Wvb = (unsigned short*)(ws + 16 * MB);
  unsigned short* Wob = (unsigned short*)(ws + 18 * MB);
  unsigned short* Qb  = (unsigned short*)(ws + 20 * MB);
  unsigned short* Kb  = (unsigned short*)(ws + 24 * MB);
  unsigned short* Vt  = (unsigned short*)(ws + 28 * MB);
  unsigned short* cxb = Xqb;   // ctx bf16 reuses the dead Xq-bf16 region

  const int nX4 = SEQ * DM / 4;
  const int nW4 = DM * DM / 4;
  cvt_f32_bf16<<<nX4 / 256, 256, 0, stream>>>(Xq, Xqb, nX4);
  cvt_f32_bf16<<<nX4 / 256, 256, 0, stream>>>(Xk, Xkb, nX4);
  cvt_f32_bf16<<<nX4 / 256, 256, 0, stream>>>(Xv, Xvb, nX4);
  cvt_f32_bf16<<<nW4 / 256, 256, 0, stream>>>(Wq, Wqb, nW4);
  cvt_f32_bf16<<<nW4 / 256, 256, 0, stream>>>(Wk, Wkb, nW4);
  cvt_f32_bf16<<<nW4 / 256, 256, 0, stream>>>(Wv, Wvb, nW4);
  cvt_f32_bf16<<<nW4 / 256, 256, 0, stream>>>(Wo, Wob, nW4);

  const int gemmBlocks = (SEQ / 128) * (DM / 64);          // 256 blocks of 8 waves
  gemm_bf16_xwT<<<gemmBlocks, 256, 0, stream>>>(Xqb, Wqb, bq, (void*)Qb, 0);
  gemm_bf16_xwT<<<gemmBlocks, 256, 0, stream>>>(Xkb, Wkb, bk, (void*)Kb, 0);
  gemm_bf16_xwT<<<gemmBlocks, 256, 0, stream>>>(Xvb, Wvb, bv, (void*)Vt, 1);

  attn_flash<<<(SEQ / 16) * (NH / 4), 128, 0, stream>>>(Qb, Kb, Vt, cxb);

  gemm_bf16_xwT<<<gemmBlocks, 256, 0, stream>>>(cxb, Wob, bo, d_out, 2);
}
